// QLayer_Block_61048665145806
// MI455X (gfx1250) — 8-bit matrix operand order repaired and checked by reading, compiled, not run on hardware
//
#include <hip/hip_runtime.h>
#include <stdint.h>


#define B_  128
#define N_  196
#define D_  768
#define H_  3072
#define R_  384
#define BN_ (B_ * N_)

#define ASZ (128 * 160)
#define BSZ (128 * 80)

typedef int v8i __attribute__((ext_vector_type(8)));
typedef unsigned int u32x4 __attribute__((ext_vector_type(4)));
typedef int i32x8v __attribute__((ext_vector_type(8)));
typedef int i32x4v __attribute__((ext_vector_type(4)));

#ifndef USE_TDM
#  if defined(__has_builtin)
#    if __has_builtin(__builtin_amdgcn_tensor_load_to_lds)
#      define USE_TDM 1
#    endif
#  endif
#endif
#ifndef USE_TDM
#  define USE_TDM 0
#endif

__device__ __forceinline__ float clamp_q(float v) {
  return fminf(127.0f, fmaxf(-128.0f, v));
}
__device__ __forceinline__ float fq_act(float v, float s) {
  return s * clamp_q(rintf(v / s));
}
__device__ __forceinline__ int8_t q8(float v, float s) {
  return (int8_t)(int)clamp_q(rintf(v / s));
}
__device__ __forceinline__ float fq_wv(float v, float s) {
  return s * clamp_q(rintf(v / s));
}

#if USE_TDM
__device__ __forceinline__ void tdm_load_2d(unsigned ldsOff, const void* gptr,
                                            unsigned tileX, unsigned tileY,
                                            unsigned tensorX, unsigned tensorY,
                                            unsigned strideX) {
  unsigned long long ga = (unsigned long long)(size_t)gptr;
  u32x4 g0;
  g0[0] = 1u;
  g0[1] = ldsOff;
  g0[2] = (unsigned)(ga & 0xFFFFFFFFu);
  g0[3] = (unsigned)((ga >> 32) & 0x01FFFFFFu) | (2u << 30);
  i32x8v g1;
  g1[0] = (int)((1u << 20) | (3u << 22) | (3u << 25));
  g1[1] = (int)((tensorX & 0xFFFFu) << 16);
  g1[2] = (int)(((tensorX >> 16) & 0xFFFFu) | ((tensorY & 0xFFFFu) << 16));
  g1[3] = (int)(((tensorY >> 16) & 0xFFFFu) | ((tileX & 0xFFFFu) << 16));
  g1[4] = (int)(tileY & 0xFFFFu);
  g1[5] = (int)strideX;
  g1[6] = 0;
  g1[7] = 0;
  i32x4v z4 = {0, 0, 0, 0};
  i32x8v z8 = {0, 0, 0, 0, 0, 0, 0, 0};
  __builtin_amdgcn_tensor_load_to_lds(g0, g1, z4, z4, z8, 0);
}
#endif

__global__ void zero_i32_kernel(int* p, int n) {
  if ((int)threadIdx.x < n) p[threadIdx.x] = 0;
}

__global__ void absmax_kernel(const float* __restrict__ src, int n, int* __restrict__ target) {
  float m = 0.0f;
  for (long i = (long)blockIdx.x * blockDim.x + threadIdx.x; i < n;
       i += (long)gridDim.x * blockDim.x)
    m = fmaxf(m, fabsf(src[i]));
  atomicMax(target, __float_as_int(m));
}

__global__ void finalize_scales_kernel(const int* __restrict__ absmax, float* __restrict__ scales) {
  int i = threadIdx.x;
  if (i < 16) scales[i] = __int_as_float(absmax[i]) / 127.0f + 1e-8f;
}

__global__ void quant_attn_kernel(const float* __restrict__ w, int8_t* __restrict__ out,
                                  const float* __restrict__ sPtr) {
  int m = blockIdx.x;
  int k = threadIdx.x;
  float s = *sPtr;
  int8_t q = 0;
  if (k < N_) q = q8(w[m * N_ + k], s);
  out[m * 256 + k] = q;
}

__global__ void quant_transpose_kernel(const float* __restrict__ W, int8_t* __restrict__ WT,
                                       int O, int I, const float* __restrict__ sPtr) {
  __shared__ float tile[32][33];
  int i0 = blockIdx.x * 32;
  int o0 = blockIdx.y * 32;
#pragma unroll
  for (int r = 0; r < 4; ++r) {
    int o = o0 + threadIdx.y + r * 8;
    int i = i0 + threadIdx.x;
    tile[threadIdx.y + r * 8][threadIdx.x] = (o < O && i < I) ? W[(long)o * I + i] : 0.0f;
  }
  __syncthreads();
  float s = *sPtr;
#pragma unroll
  for (int r = 0; r < 4; ++r) {
    int i = i0 + threadIdx.y + r * 8;
    int o = o0 + threadIdx.x;
    if (i < I && o < O) WT[(long)i * O + o] = q8(tile[threadIdx.x][threadIdx.y + r * 8], s);
  }
}

__global__ void norm1_quant_kernel(const float* __restrict__ x, const float* __restrict__ a,
                                   const float* __restrict__ b, const float* __restrict__ scl,
                                   const float* __restrict__ actS, int8_t* __restrict__ xq) {
  long idx = (long)blockIdx.x * blockDim.x + threadIdx.x;
  if (idx >= (long)BN_ * D_) return;
  int d = (int)(idx % D_);
  float sa = scl[0];
  float h = x[idx] * fq_wv(a[d], sa) + b[d];
  xq[idx] = q8(h, actS[0]);
}

struct GemmParams {
  const int8_t* A;
  const int8_t* B;
  int M, N, Kpad, Kact, lda, ldb;
  long bBatchStride;
  int zRowStride;
  const float* scalesDev;
  const float* actS;
  const float* bias;
  const float* xOrg;
  const float* gamma1;
  const float* norm2a;
  const float* norm2b;
  int8_t* x1q;
  int8_t* x2q;
  const float* gamma2;
  const int8_t* x1qIn;
  float* outF;
  int8_t* out8;
};

template <int EPI>
__device__ __forceinline__ void epilogue(int row, long gRow, int col, int acc,
                                         const GemmParams& p) {
  const float* S = p.actS;
  const float* W = p.scalesDev;
  float a = (float)acc;
  if (EPI == 1) {
    float f = a * (W[1] * S[0]) + p.bias[row];
    f = fq_act(f, S[1]);
    f *= fq_wv(p.gamma1[col], W[2]);
    f += p.xOrg[gRow * D_ + col];
    float q = clamp_q(rintf(f / S[8]));
    p.x1q[gRow * D_ + col] = (int8_t)(int)q;
    float x1 = S[8] * q;
    float h2 = x1 * fq_wv(p.norm2a[col], W[3]) + p.norm2b[col];
    p.x2q[gRow * D_ + col] = q8(h2, S[2]);
  } else if (EPI == 2) {
    float f = a * (W[4] * S[2]) + p.bias[col];
    p.out8[gRow * (long)p.N + col] = q8(f, S[4]);
  } else if (EPI == 3) {
    float f = a * (W[5] * S[4]) + p.bias[col];
    f = fq_act(f, S[5]);
    float g = 0.5f * f * (1.0f + erff(f * 0.70710678118654752f));
    p.out8[gRow * (long)p.N + col] = q8(g, S[6]);
  } else if (EPI == 4) {
    float f = a * (W[6] * S[6]) + p.bias[col];
    p.out8[gRow * (long)p.N + col] = q8(f, S[7]);
  } else {
    float f = a * (W[7] * S[7]) + p.bias[col];
    f = fq_act(f, S[3]);
    f *= fq_wv(p.gamma2[col], W[8]);
    f += S[8] * (float)p.x1qIn[gRow * D_ + col];
    p.outF[gRow * D_ + col] = fq_act(f, S[9]);
  }
}

template <int EPI, bool GUARD>
__global__ __launch_bounds__(256) void gemm_i8_wmma(GemmParams p) {
#if USE_TDM
  __shared__ alignas(16) int8_t As[2 * ASZ];
  __shared__ alignas(16) int8_t Bs[2 * BSZ];
#else
  __shared__ alignas(16) int8_t As[ASZ];
  __shared__ alignas(16) int8_t Bs[BSZ];
#endif

  const int t = threadIdx.x;
  const int lane = t & 31;
  const int wave = t >> 5;
  const int wm = (wave & 3) * 32;
  const int wn = (wave >> 2) * 32;
  const int m0 = blockIdx.y * 128;
  const int n0 = blockIdx.x * 64;
  const int z = blockIdx.z;

  const int8_t* __restrict__ Ab = p.A;
  const int8_t* __restrict__ Bb = p.B + (long)z * p.bBatchStride;

  v8i acc[2][2];
#pragma unroll
  for (int i = 0; i < 2; ++i)
#pragma unroll
    for (int j = 0; j < 2; ++j)
#pragma unroll
      for (int e = 0; e < 8; ++e) acc[i][j][e] = 0;

  const int h = lane >> 4;
  const int ml = lane & 15;

#if USE_TDM
  const unsigned asOff = (unsigned)(size_t)(void*)As;
  const unsigned bsOff = (unsigned)(size_t)(void*)Bs;
  const int nIter = p.Kpad >> 7;

  if (wave == 0) {
    tdm_load_2d(asOff, Ab + (long)m0 * p.lda, 128u, 128u,
                (unsigned)p.lda, (unsigned)(p.M - m0), (unsigned)p.lda);
    tdm_load_2d(bsOff, Bb + n0, 64u, 128u,
                (unsigned)(p.ldb - n0), (unsigned)p.Kact, (unsigned)p.ldb);
  }

  for (int it = 0; it < nIter; ++it) {
    const int kt = it << 7;
    const int buf = it & 1;
    if (wave == 0) __builtin_amdgcn_s_wait_tensorcnt(0);
    __syncthreads();
    if (wave == 0 && (it + 1) < nIter) {
      const int nk = kt + 128;
      const int nb = buf ^ 1;
      tdm_load_2d(asOff + (unsigned)(nb * ASZ), Ab + (long)m0 * p.lda + nk, 128u, 128u,
                  (unsigned)(p.lda - nk), (unsigned)(p.M - m0), (unsigned)p.lda);
      tdm_load_2d(bsOff + (unsigned)(nb * BSZ), Bb + (long)nk * p.ldb + n0, 64u, 128u,
                  (unsigned)(p.ldb - n0), (unsigned)(p.Kact - nk), (unsigned)p.ldb);
    }
    const int8_t* __restrict__ AsB = As + buf * ASZ;
    const int8_t* __restrict__ BsB = Bs + buf * BSZ;
#else
  for (int kt = 0; kt < p.Kpad; kt += 128) {
    {
      const int rowF = t >> 1;
      const int halfA = (t & 1) * 64;
      const int halfB = (t & 1) * 32;
      int4 av[4], bv[2];
#pragma unroll
      for (int c = 0; c < 4; ++c) av[c] = make_int4(0, 0, 0, 0);
      bv[0] = bv[1] = make_int4(0, 0, 0, 0);
      if (!GUARD || (m0 + rowF) < p.M) {
        const int4* srcA =
            reinterpret_cast<const int4*>(Ab + (long)(m0 + rowF) * p.lda + kt + halfA);
#pragma unroll
        for (int c = 0; c < 4; ++c) av[c] = srcA[c];
      }
      if (!GUARD || (kt + rowF) < p.Kact) {
        const int4* srcB =
            reinterpret_cast<const int4*>(Bb + (long)(kt + rowF) * p.ldb + n0 + halfB);
        bv[0] = srcB[0];
        bv[1] = srcB[1];
      }
      int aBase = rowF * 160 + (halfA ? 80 : 0);
#pragma unroll
      for (int c = 0; c < 4; ++c)
        *reinterpret_cast<int4*>(&As[aBase + c * 16]) = av[c];
      *reinterpret_cast<int4*>(&Bs[rowF * 80 + halfB]) = bv[0];
      *reinterpret_cast<int4*>(&Bs[rowF * 80 + halfB + 16]) = bv[1];
      if (kt + 128 < p.Kpad) {
        __builtin_prefetch(Ab + (long)(m0 + rowF) * p.lda + (kt + 128) + halfA, 0, 3);
        __builtin_prefetch(Bb + (long)(kt + 128 + rowF) * p.ldb + n0 + halfB, 0, 3);
      }
    }
    __syncthreads();
    const int8_t* __restrict__ AsB = As;
    const int8_t* __restrict__ BsB = Bs;
#endif

#pragma unroll
    for (int ks = 0; ks < 2; ++ks) {
      v8i afrag[2], bfrag[2];
#pragma unroll
      for (int tm = 0; tm < 2; ++tm) {
        const int8_t* base = &AsB[(wm + tm * 16 + ml) * 160 + ks * 80 + h * 8];
        int2 r0 = *reinterpret_cast<const int2*>(base + 0);
        int2 r1 = *reinterpret_cast<const int2*>(base + 16);
        int2 r2 = *reinterpret_cast<const int2*>(base + 32);
        int2 r3 = *reinterpret_cast<const int2*>(base + 48);
        afrag[tm][0] = r0.x; afrag[tm][1] = r0.y;
        afrag[tm][2] = r1.x; afrag[tm][3] = r1.y;
        afrag[tm][4] = r2.x; afrag[tm][5] = r2.y;
        afrag[tm][6] = r3.x; afrag[tm][7] = r3.y;
      }
#pragma unroll
      for (int tn = 0; tn < 2; ++tn) {
#pragma unroll
        for (int v = 0; v < 8; ++v) {
          int kr = ks * 64 + (v >> 1) * 16 + h * 8 + (v & 1) * 4;
          unsigned w = 0;
#pragma unroll
          for (int b = 0; b < 4; ++b)
            w |= (unsigned)(uint8_t)BsB[(kr + b) * 80 + wn + tn * 16 + ml] << (8 * b);
          bfrag[tn][v] = (int)w;
        }
      }
#pragma unroll
      for (int tm = 0; tm < 2; ++tm)
#pragma unroll
        for (int tn = 0; tn < 2; ++tn)
          acc[tm][tn] = __builtin_amdgcn_wmma_i32_16x16x64_iu8(
              true, afrag[tm], true, bfrag[tn], acc[tm][tn], false, false);
    }
#if !USE_TDM
    __syncthreads();
#endif
  }

#pragma unroll
  for (int tm = 0; tm < 2; ++tm) {
#pragma unroll
    for (int tn = 0; tn < 2; ++tn) {
#pragma unroll
      for (int r = 0; r < 8; ++r) {
        int mrow = m0 + wm + tm * 16 + h * 8 + r;
        if (GUARD && mrow >= p.M) continue;
        int col = n0 + wn + tn * 16 + ml;
        long gRow = (long)z * p.zRowStride + mrow;
        epilogue<EPI>(mrow, gRow, col, acc[tm][tn][r], p);
      }
    }
  }
}

extern "C" void kernel_launch(void* const* d_in, const int* in_sizes, int n_in,
                              void* d_out, int out_size, void* d_ws, size_t ws_size,
                              hipStream_t stream) {
  const float* x        = (const float*)d_in[0];
  const float* norm1_a  = (const float*)d_in[1];
  const float* norm1_b  = (const float*)d_in[2];
  const float* attn_w   = (const float*)d_in[3];
  const float* attn_b   = (const float*)d_in[4];
  const float* gamma1   = (const float*)d_in[5];
  const float* norm2_a  = (const float*)d_in[6];
  const float* norm2_b  = (const float*)d_in[7];
  const float* fc1_vt_w = (const float*)d_in[8];
  const float* fc1_vt_b = (const float*)d_in[9];
  const float* fc1_u_w  = (const float*)d_in[10];
  const float* fc1_u_b  = (const float*)d_in[11];
  const float* fc2_vt_w = (const float*)d_in[12];
  const float* fc2_vt_b = (const float*)d_in[13];
  const float* fc2_u_w  = (const float*)d_in[14];
  const float* fc2_u_b  = (const float*)d_in[15];
  const float* gamma2   = (const float*)d_in[16];
  const float* actS     = (const float*)d_in[17];

  uint8_t* ws = (uint8_t*)d_ws;
  size_t off = 0;
  auto alloc = [&](size_t n) {
    off = (off + 255) & ~(size_t)255;
    size_t o = off;
    off += n;
    return o;
  };
  int*    absmax  = (int*)(ws + alloc(64));
  float*  scl     = (float*)(ws + alloc(64));
  int8_t* attn_wq = (int8_t*)(ws + alloc((size_t)N_ * 256));
  int8_t* w2T     = (int8_t*)(ws + alloc((size_t)D_ * R_));
  int8_t* w3T     = (int8_t*)(ws + alloc((size_t)R_ * H_));
  int8_t* w4T     = (int8_t*)(ws + alloc((size_t)H_ * R_));
  int8_t* w5T     = (int8_t*)(ws + alloc((size_t)R_ * D_));
  int8_t* xq      = (int8_t*)(ws + alloc((size_t)BN_ * D_));
  int8_t* x2q     = (int8_t*)(ws + alloc((size_t)BN_ * D_));
  int8_t* x1q     = (int8_t*)(ws + alloc((size_t)BN_ * D_));
  int8_t* h3q     = (int8_t*)(ws + alloc((size_t)BN_ * R_));
  int8_t* h4q     = (int8_t*)(ws + alloc((size_t)BN_ * H_));
  int8_t* h5q     = (int8_t*)(ws + alloc((size_t)BN_ * R_));
  (void)ws_size; (void)in_sizes; (void)n_in; (void)out_size;

  zero_i32_kernel<<<1, 32, 0, stream>>>(absmax, 16);
  auto amax = [&](const float* src, int n, int slot) {
    int blocks = (n + 255) / 256;
    if (blocks > 512) blocks = 512;
    absmax_kernel<<<blocks, 256, 0, stream>>>(src, n, absmax + slot);
  };
  amax(norm1_a, D_, 0);
  amax(attn_w, N_ * N_, 1);
  amax(gamma1, D_, 2);
  amax(norm2_a, D_, 3);
  amax(fc1_vt_w, R_ * D_, 4);
  amax(fc1_u_w, H_ * R_, 5);
  amax(fc2_vt_w, R_ * H_, 6);
  amax(fc2_u_w, D_ * R_, 7);
  amax(gamma2, D_, 8);
  finalize_scales_kernel<<<1, 32, 0, stream>>>(absmax, scl);

  quant_attn_kernel<<<N_, 256, 0, stream>>>(attn_w, attn_wq, scl + 1);
  {
    dim3 b(32, 8);
    quant_transpose_kernel<<<dim3(D_ / 32, R_ / 32), b, 0, stream>>>(fc1_vt_w, w2T, R_, D_, scl + 4);
    quant_transpose_kernel<<<dim3(R_ / 32, H_ / 32), b, 0, stream>>>(fc1_u_w, w3T, H_, R_, scl + 5);
    quant_transpose_kernel<<<dim3(H_ / 32, R_ / 32), b, 0, stream>>>(fc2_vt_w, w4T, R_, H_, scl + 6);
    quant_transpose_kernel<<<dim3(R_ / 32, D_ / 32), b, 0, stream>>>(fc2_u_w, w5T, D_, R_, scl + 7);
  }

  norm1_quant_kernel<<<(BN_ * D_) / 256, 256, 0, stream>>>(x, norm1_a, norm1_b, scl, actS, xq);

  GemmParams p{};
  p.scalesDev = scl;
  p.actS = actS;
  p.zRowStride = N_;

  p.A = attn_wq; p.lda = 256; p.M = N_; p.Kpad = 256; p.Kact = N_;
  p.B = xq; p.ldb = D_; p.N = D_; p.bBatchStride = (long)N_ * D_;
  p.bias = attn_b; p.xOrg = x; p.gamma1 = gamma1; p.norm2a = norm2_a; p.norm2b = norm2_b;
  p.x1q = x1q; p.x2q = x2q;
  gemm_i8_wmma<1, true><<<dim3(D_ / 64, 2, B_), 256, 0, stream>>>(p);

  p = GemmParams{};
  p.scalesDev = scl; p.actS = actS; p.zRowStride = 0;
  p.A = x2q; p.lda = D_; p.M = BN_; p.Kpad = D_; p.Kact = D_;
  p.B = w2T; p.ldb = R_; p.N = R_; p.bias = fc1_vt_b; p.out8 = h3q;
  gemm_i8_wmma<2, false><<<dim3(R_ / 64, BN_ / 128, 1), 256, 0, stream>>>(p);

  p.A = h3q; p.lda = R_; p.Kpad = R_; p.Kact = R_;
  p.B = w3T; p.ldb = H_; p.N = H_; p.bias = fc1_u_b; p.out8 = h4q;
  gemm_i8_wmma<3, false><<<dim3(H_ / 64, BN_ / 128, 1), 256, 0, stream>>>(p);

  p.A = h4q; p.lda = H_; p.Kpad = H_; p.Kact = H_;
  p.B = w4T; p.ldb = R_; p.N = R_; p.bias = fc2_vt_b; p.out8 = h5q;
  gemm_i8_wmma<4, false><<<dim3(R_ / 64, BN_ / 128, 1), 256, 0, stream>>>(p);

  p.A = h5q; p.lda = R_; p.Kpad = R_; p.Kact = R_;
  p.B = w5T; p.ldb = D_; p.N = D_; p.bias = fc2_u_b;
  p.gamma2 = gamma2; p.x1qIn = x1q; p.outF = (float*)d_out;
  gemm_i8_wmma<5, false><<<dim3(D_ / 64, BN_ / 128, 1), 256, 0, stream>>>(p);
}
